// ParaEncoderGRAN_79190607004333
// MI455X (gfx1250) — matrix operand order repaired and checked by reading, not compiled, not run on hardware
//
#include <hip/hip_runtime.h>
#include <hip/hip_bf16.h>
#include <math.h>

#define B_   32
#define S_   512
#define H_   512
#define G4   2048   // 4*H

typedef __bf16 bf16;
typedef float  v8f   __attribute__((ext_vector_type(8)));
typedef bf16   v16bf __attribute__((ext_vector_type(16)));
typedef bf16   v8bf  __attribute__((ext_vector_type(8)));

static __device__ __forceinline__ v8f wmma_bf16(v16bf a, v16bf b, v8f c) {
  // D = A(16x32 bf16) * B(32x16 bf16) + C(16x16 f32)
  return __builtin_amdgcn_wmma_f32_16x16x32_bf16(false, a, false, b, (short)0, c,
                                                 false, false);
}
static __device__ __forceinline__ v16bf cat16(v8bf lo, v8bf hi) {
  return __builtin_shufflevector(lo, hi, 0,1,2,3,4,5,6,7,8,9,10,11,12,13,14,15);
}
// A tile 16x32 bf16 from row-major src (leading dim ld elements), origin p=[row0][k0].
// Lane layout: lanes 0-15 hold K {0..7,16..23}, lanes 16-31 hold K {8..15,24..31}.
static __device__ __forceinline__ v16bf load_a_bf16(const bf16* p, int ld, int lane) {
  int row = lane & 15, kh = (lane >> 4) & 1;
  const bf16* q = p + row * ld + kh * 8;
  return cat16(*(const v8bf*)q, *(const v8bf*)(q + 16));
}
// B tile 32x16 from W stored [col][k] row-major in k (ldk elements), k0 = tile K origin.
// Lane layout: lanes 0-15 hold K {0..7,16..23} of col=lane, lanes 16-31 hold K {8..15,24..31}.
static __device__ __forceinline__ v16bf load_b_bf16(const bf16* W, int ldk, int k0, int lane) {
  int col = lane & 15, kh = (lane >> 4) & 1;
  const bf16* q = W + (long)col * ldk + k0 + kh * 8;
  return cat16(*(const v8bf*)q, *(const v8bf*)(q + 16));
}
static __device__ __forceinline__ float sigf(float x) { return 1.f / (1.f + __expf(-x)); }

// ---------------------------------------------------------------- utility kernels
__global__ void k_zero(float* p, long n) {
  for (long i = blockIdx.x * (long)blockDim.x + threadIdx.x; i < n;
       i += (long)gridDim.x * blockDim.x) p[i] = 0.f;
}
__global__ void k_cvt(const float* __restrict__ in, bf16* __restrict__ out, int n, float sc) {
  for (int i = blockIdx.x * blockDim.x + threadIdx.x; i < n; i += gridDim.x * blockDim.x)
    out[i] = (bf16)(in[i] * sc);
}
// embedding gather -> f32 copy + bf16 copy
__global__ void k_embed(const int* __restrict__ tok, const float* __restrict__ emb,
                        float* __restrict__ ef, bf16* __restrict__ eb) {
  long bs = blockIdx.x;
  long t = tok[bs];
  const float* src = emb + t * H_;
  float* d  = ef + bs * H_;
  bf16*  db = eb + bs * H_;
  for (int h = threadIdx.x; h < H_; h += blockDim.x) {
    float v = src[h]; d[h] = v; db[h] = (bf16)v;
  }
}

// ------------------------------------------- Pre = emb @ Wih^T + (bih+bhh), WMMA
// A: [16384][512] bf16, W: [2048][512] bf16, out: [16384][2048] f32
// All 8 waves of a block share the same 16 weight columns -> stage them in LDS once.
__global__ __launch_bounds__(256) void k_pre_gemm(const bf16* __restrict__ A,
                                                  const bf16* __restrict__ W,
                                                  const float* __restrict__ b0,
                                                  const float* __restrict__ b1,
                                                  float* __restrict__ out) {
  __shared__ bf16 wlds[16 * H_];           // 16 KB weight slice
  int tid = threadIdx.x;
  int lane = tid & 31, wv = tid >> 5;
  long tileBase = (long)blockIdx.x * 8;    // 131072 16x16 tiles total
  int nt = (int)(tileBase >> 10);          // 0..127 (uniform across block)
  int mt = (int)(tileBase & 1023) + wv;    // 0..1023

  const bf16* Wbase = W + (long)(nt * 16) * H_;
  for (int i = tid; i < (16 * H_) / 8; i += 256)
    ((v8bf*)wlds)[i] = ((const v8bf*)Wbase)[i];
  __syncthreads();

  int col = nt * 16 + (lane & 15);
  const bf16* Arow = A + (long)(mt * 16) * H_;
  v8f acc = {};
#pragma unroll
  for (int kt = 0; kt < 16; ++kt) {
    v16bf a = load_a_bf16(Arow + kt * 32, H_, lane);
    v16bf b = load_b_bf16(wlds, H_, kt * 32, lane);
    acc = wmma_bf16(a, b, acc);
  }
  float bias = b0[col] + b1[col];
  int rbase = ((lane >> 4) & 1) * 8;
#pragma unroll
  for (int r = 0; r < 8; ++r)
    out[(long)(mt * 16 + rbase + r) * G4 + col] = acc[r] + bias;
}

// ------------------------------------------------------- recurrent LSTM scan
// grid = 32 workgroups: blockIdx[0..15] = forward (owning 32 hidden units each),
// blockIdx[16..31] = backward. Per-step device-wide barrier via atomics.
__global__ __launch_bounds__(256) void k_scan(const float* __restrict__ PreF,
                                              const float* __restrict__ PreB,
                                              const bf16* __restrict__ WhhFb,
                                              const bf16* __restrict__ WhhBb,
                                              const int* __restrict__ lens,
                                              bf16* __restrict__ hbuf,   // [2][32][512]
                                              float* __restrict__ out,   // [32][512][1024]
                                              unsigned* __restrict__ bar) {
  __shared__ bf16  hlds[B_][H_];           // full h_prev (all units), bf16
  __shared__ float glds[4 * 32 * 33];      // [gate][unit_local][batch], padded
  __shared__ float cst[32 * 32];           // cell state slice [unit_local][batch]
  __shared__ int   llds[B_];

  int dir = blockIdx.x >> 4;
  int wg  = blockIdx.x & 15;
  int tid = threadIdx.x;
  int lane = tid & 31, wv = tid >> 5;
  const float* Pre = dir ? PreB : PreF;
  const bf16*  Whh = dir ? WhhBb : WhhFb;
  bf16* hb = hbuf + dir * (B_ * H_);

  for (int i = tid; i < 32 * 32; i += 256) cst[i] = 0.f;
  if (tid < B_) llds[tid] = lens[tid];

  // Preload this wave's 16-column slice of Whh^T into registers (128 VGPRs),
  // reused across all 512 time steps.
  int gi = wv >> 1;                                  // gate group i,f,g,o
  int colbase = gi * H_ + wg * 32 + (wv & 1) * 16;   // global gate column base
  int col = colbase + (lane & 15);
  int kh = (lane >> 4) & 1;
  v16bf wb[16];
#pragma unroll
  for (int kt = 0; kt < 16; ++kt) {
    const bf16* q = Whh + (long)col * H_ + kt * 32 + kh * 8;
    wb[kt] = cat16(*(const v8bf*)q, *(const v8bf*)(q + 16));
  }
  __syncthreads();

  for (int t = 0; t < S_; ++t) {
    { // pull full h_prev into LDS (32 KB, b128 copies)
      const v8bf* src = (const v8bf*)hb;
      v8bf* dst = (v8bf*)&hlds[0][0];
      for (int i = tid; i < (B_ * H_) / 8; i += 256) dst[i] = src[i];
    }
    // prefetch next step's pre-activation segments (4 gate groups x 32 batches;
    // each segment = 32 floats = one 128B line) while this step computes
    if (t + 1 < S_ && tid < 128) {
      int pb = tid & 31;
      int pg = tid >> 5;
      int tn = dir ? max(llds[pb] - 2 - t, 0) : (t + 1);
      __builtin_prefetch(Pre + ((long)pb * S_ + tn) * G4 + pg * H_ + wg * 32, 0, 2);
    }
    __syncthreads();

    v8f acc0 = {}, acc1 = {};                 // batch rows 0-15 / 16-31
#pragma unroll
    for (int kt = 0; kt < 16; ++kt) {
      v16bf A0 = load_a_bf16(&hlds[0][0]  + kt * 32, H_, lane);
      v16bf A1 = load_a_bf16(&hlds[16][0] + kt * 32, H_, lane);
      acc0 = wmma_bf16(A0, wb[kt], acc0);
      acc1 = wmma_bf16(A1, wb[kt], acc1);
    }

    int ul = (wv & 1) * 16 + (lane & 15);     // local column within gate group
#pragma unroll
    for (int r = 0; r < 8; ++r) {
      int b0i = r + 8 * kh;
      int b1i = b0i + 16;
      int t0 = dir ? max(llds[b0i] - 1 - t, 0) : t;
      int t1 = dir ? max(llds[b1i] - 1 - t, 0) : t;
      glds[(gi * 32 + ul) * 33 + b0i] = acc0[r] + Pre[((long)b0i * S_ + t0) * G4 + col];
      glds[(gi * 32 + ul) * 33 + b1i] = acc1[r] + Pre[((long)b1i * S_ + t1) * G4 + col];
    }
    __syncthreads();

    // element-wise LSTM cell update for this wg's 32 units x 32 batches
    for (int c = tid; c < 1024; c += 256) {
      int u = c >> 5, b = c & 31;
      float iv = glds[(0 * 32 + u) * 33 + b];
      float fv = glds[(1 * 32 + u) * 33 + b];
      float gv = glds[(2 * 32 + u) * 33 + b];
      float ov = glds[(3 * 32 + u) * 33 + b];
      float cp = cst[u * 32 + b];
      float cn = sigf(fv) * cp + sigf(iv) * tanhf(gv);
      float hn = sigf(ov) * tanhf(cn);
      int ug = wg * 32 + u;
      if (t < llds[b]) {
        cst[u * 32 + b] = cn;
        hb[b * H_ + ug] = (bf16)hn;
        if (dir) {
          int s = llds[b] - 1 - t;
          out[((long)b * S_ + s) * 1024 + 512 + ug] = hn;
        } else {
          out[((long)b * S_ + t) * 1024 + ug] = hn;
        }
      } else if (!dir) {
        out[((long)b * S_ + t) * 1024 + ug] = 0.f;
      }
    }

    // device-wide step barrier: release fence by ALL threads (their stores),
    // arrival by thread 0, acquire fence by ALL threads (their next-step loads).
    __threadfence();
    __syncthreads();
    if (tid == 0) {
      atomicAdd(bar, 1u);
      unsigned tgt = 32u * (t + 1);
      while (__hip_atomic_load(bar, __ATOMIC_RELAXED, __HIP_MEMORY_SCOPE_AGENT) < tgt)
        __builtin_amdgcn_s_sleep(1);
    }
    __syncthreads();
    __threadfence();
  }
}

// -------------------------- gate = sigmoid(2*Wx@e + Wh@(of+ob) + b); mean-pool e*gate
// grid: 32(b) x 32(ntile) x 8(s-chunk of 64) = 8192 wgs, 128 threads (4 waves).
// The block's 16 columns of (2*Wx) and Wh are staged in LDS once.
__global__ __launch_bounds__(128) void k_gate(const bf16* __restrict__ eb,
                                              const float* __restrict__ ef,
                                              const float* __restrict__ lstm, // [B][S][1024]
                                              const bf16* __restrict__ Wx2b,  // holds 2*Wx
                                              const bf16* __restrict__ Whb,
                                              const float* __restrict__ bg,
                                              float* __restrict__ partial) {  // [B][8][512]
  __shared__ bf16 wxl[16 * H_];
  __shared__ bf16 whl[16 * H_];
  __shared__ float red[4][16];
  int lane = threadIdx.x & 31, wv = threadIdx.x >> 5;
  int x = blockIdx.x;
  int b  = x >> 8;
  int nt = (x >> 3) & 31;                  // uniform across block
  int sc = x & 7;

  const bf16* WxB = Wx2b + (long)(nt * 16) * H_;
  const bf16* WhB = Whb  + (long)(nt * 16) * H_;
  for (int i = threadIdx.x; i < (16 * H_) / 8; i += 128) {
    ((v8bf*)wxl)[i] = ((const v8bf*)WxB)[i];
    ((v8bf*)whl)[i] = ((const v8bf*)WhB)[i];
  }
  __syncthreads();

  int sbase = sc * 64 + wv * 16;
  int col = nt * 16 + (lane & 15);
  int kh = (lane >> 4) & 1;
  int arow = lane & 15;
  const float* orow = lstm + ((long)(b * S_) + sbase + arow) * 1024;
  v8f acc = {};
#pragma unroll 4
  for (int kt = 0; kt < 16; ++kt) {
    v16bf a1 = load_a_bf16(eb + (long)(b * S_ + sbase) * H_ + kt * 32, H_, lane);
    v16bf b1 = load_b_bf16(wxl, H_, kt * 32, lane);
    acc = wmma_bf16(a1, b1, acc);
    // A2 = out_f + out_b in bf16, built on the fly
    int k0 = kt * 32 + kh * 8;
    v8bf lo, hi;
#pragma unroll
    for (int i = 0; i < 8; ++i) {
      lo[i] = (bf16)(orow[k0 + i]      + orow[512 + k0 + i]);
      hi[i] = (bf16)(orow[k0 + 16 + i] + orow[512 + k0 + 16 + i]);
    }
    v16bf a2 = cat16(lo, hi);
    v16bf b2 = load_b_bf16(whl, H_, kt * 32, lane);
    acc = wmma_bf16(a2, b2, acc);
  }
  float bgc = bg[col];
  float psum = 0.f;
#pragma unroll
  for (int r = 0; r < 8; ++r) {
    int srow = sbase + r + 8 * kh;
    float g = sigf(acc[r] + bgc);
    psum += ef[((long)(b * S_) + srow) * H_ + col] * g;
  }
  psum += __shfl_xor(psum, 16);            // fold rows +8 (same column)
  if (lane < 16) red[wv][lane] = psum;
  __syncthreads();
  if (threadIdx.x < 16) {
    float s = red[0][threadIdx.x] + red[1][threadIdx.x] + red[2][threadIdx.x] + red[3][threadIdx.x];
    partial[((long)b * 8 + sc) * H_ + nt * 16 + threadIdx.x] = s;
  }
}

__global__ void k_hf(const float* __restrict__ partial, float* __restrict__ hf) {
  int i = blockIdx.x * blockDim.x + threadIdx.x;   // 16384
  int b = i >> 9, u = i & 511;
  float s = 0.f;
#pragma unroll
  for (int c = 0; c < 8; ++c) s += partial[((long)b * 8 + c) * H_ + u];
  s *= (1.f / 512.f);
  hf[b * 1024 + u] = s;
  hf[b * 1024 + 512 + u] = s;
}

// ---------------------------------------------------------------- host launcher
extern "C" void kernel_launch(void* const* d_in, const int* in_sizes, int n_in,
                              void* d_out, int out_size, void* d_ws, size_t ws_size,
                              hipStream_t stream) {
  const int*   tokens = (const int*)d_in[0];
  const int*   lens   = (const int*)d_in[1];
  const float* emb    = (const float*)d_in[2];
  const float* Wih_f  = (const float*)d_in[3];
  const float* Whh_f  = (const float*)d_in[4];
  const float* bih_f  = (const float*)d_in[5];
  const float* bhh_f  = (const float*)d_in[6];
  const float* Wih_b  = (const float*)d_in[7];
  const float* Whh_b  = (const float*)d_in[8];
  const float* bih_b  = (const float*)d_in[9];
  const float* bhh_b  = (const float*)d_in[10];
  const float* Wx     = (const float*)d_in[11];
  const float* Wh     = (const float*)d_in[12];
  const float* b_gate = (const float*)d_in[13];
  float* out = (float*)d_out;
  float* hf  = out + (long)B_ * S_ * 1024;

  char* w = (char*)d_ws;
  size_t o = 0;
  unsigned* bar    = (unsigned*)(w + o); o += 256;
  bf16*  hbuf      = (bf16*) (w + o); o += 2L * B_ * H_ * sizeof(bf16);        // 64 KB
  float* partial   = (float*)(w + o); o += (long)B_ * 8 * H_ * sizeof(float);  // 512 KB
  float* ef        = (float*)(w + o); o += (long)B_ * S_ * H_ * sizeof(float); // 32 MB
  bf16*  eb        = (bf16*) (w + o); o += (long)B_ * S_ * H_ * sizeof(bf16);  // 16 MB
  float* PreF      = (float*)(w + o); o += (long)B_ * S_ * G4 * sizeof(float); // 128 MB
  float* PreB      = (float*)(w + o); o += (long)B_ * S_ * G4 * sizeof(float); // 128 MB
  bf16*  WihFb     = (bf16*) (w + o); o += (long)G4 * H_ * sizeof(bf16);
  bf16*  WihBb     = (bf16*) (w + o); o += (long)G4 * H_ * sizeof(bf16);
  bf16*  WhhFb     = (bf16*) (w + o); o += (long)G4 * H_ * sizeof(bf16);
  bf16*  WhhBb     = (bf16*) (w + o); o += (long)G4 * H_ * sizeof(bf16);
  bf16*  Wx2b      = (bf16*) (w + o); o += (long)H_ * H_ * sizeof(bf16);
  bf16*  Whb       = (bf16*) (w + o); o += (long)H_ * H_ * sizeof(bf16);
  (void)ws_size; (void)n_in; (void)in_sizes; (void)out_size;

  // zero: full output (masked backward region) + barrier + h0
  k_zero<<<4096, 256, 0, stream>>>(out, (long)B_ * S_ * 1024 + B_ * 1024);
  k_zero<<<64, 256, 0, stream>>>((float*)w, (256 + 2L * B_ * H_ * (long)sizeof(bf16)) / 4);

  // bf16 weight conversions (Wx folded with the x2 scale)
  k_cvt<<<1024, 256, 0, stream>>>(Wih_f, WihFb, G4 * H_, 1.f);
  k_cvt<<<1024, 256, 0, stream>>>(Wih_b, WihBb, G4 * H_, 1.f);
  k_cvt<<<1024, 256, 0, stream>>>(Whh_f, WhhFb, G4 * H_, 1.f);
  k_cvt<<<1024, 256, 0, stream>>>(Whh_b, WhhBb, G4 * H_, 1.f);
  k_cvt<<<256, 256, 0, stream>>>(Wx, Wx2b, H_ * H_, 2.f);
  k_cvt<<<256, 256, 0, stream>>>(Wh, Whb, H_ * H_, 1.f);

  k_embed<<<B_ * S_, 256, 0, stream>>>(tokens, emb, ef, eb);

  k_pre_gemm<<<16384, 256, 0, stream>>>(eb, WihFb, bih_f, bhh_f, PreF);
  k_pre_gemm<<<16384, 256, 0, stream>>>(eb, WihBb, bih_b, bhh_b, PreB);

  k_scan<<<32, 256, 0, stream>>>(PreF, PreB, WhhFb, WhhBb, lens, hbuf, out, bar);

  k_gate<<<8192, 128, 0, stream>>>(eb, ef, out, Wx2b, Whb, b_gate, partial);
  k_hf<<<64, 256, 0, stream>>>(partial, hf);
}
